// ChildSumTreeLSTM_38302518345966
// MI455X (gfx1250) — matrix operand order repaired and checked by reading, not compiled, not run on hardware
//
#include <hip/hip_runtime.h>
#include <hip/hip_bf16.h>

typedef __attribute__((ext_vector_type(16))) __bf16 v16bf;
typedef __attribute__((ext_vector_type(8)))  float  v8f;

#define HID 128
#define TM 16
#define LSTRIDE 132
#define NWAVE 9
#define BLOCK (NWAVE * 32)
#define TREE_DEPTH 18
#define NNODES ((1 << TREE_DEPTH) - 1)

__device__ __forceinline__ float sigmoidf_(float x) { return 1.f / (1.f + expf(-x)); }

// ---------------------------------------------------------------------------
// Pack the 8 distinct 128x128 f32 weight matrices into bf16 WMMA-B fragments.
// Matrix order: 0:Wix 1:Wox 2:Wux 3:Wfx 4:Wih 5:Woh 6:Wuh 7:Wfh
// Per matrix: 8 col-tiles (jt) x 4 k-steps (kt); block = 32 lanes x 16 halves.
// B layout (contiguous k order, same as A): lane l -> column jt*16+(l&15),
// halves h=0..15 -> k = kt*32 + (l>>4)*16 + h.  B[k][j] = W[j][k].
// ---------------------------------------------------------------------------
__global__ __launch_bounds__(256) void pack_weights_kernel(
    const float* __restrict__ Wix, const float* __restrict__ Wox,
    const float* __restrict__ Wux, const float* __restrict__ Wfx,
    const float* __restrict__ Wih, const float* __restrict__ Woh,
    const float* __restrict__ Wuh, const float* __restrict__ Wfh,
    __bf16* __restrict__ wpack) {
  const float* Ws[8] = {Wix, Wox, Wux, Wfx, Wih, Woh, Wuh, Wfh};
  int wmat = blockIdx.x;
  const float* W = Ws[wmat];
  for (int idx = threadIdx.x; idx < HID * HID; idx += blockDim.x) {
    int blockid = idx >> 9;          // (jt*4 + kt)
    int lane    = (idx >> 4) & 31;
    int h       = idx & 15;
    int jt = blockid >> 2;
    int kt = blockid & 3;
    int j = jt * 16 + (lane & 15);
    int k = kt * 32 + ((lane >> 4) * 16) + h;
    wpack[wmat * (HID * HID) + idx] = (__bf16)W[j * HID + k];
  }
}

// ---------------------------------------------------------------------------
// One launch per tree level. Each block handles 16 nodes of the level.
// 9 waves: w0..3 x-gates (Wix,Wox,Wux,Wfx), w4..6 h_sum-gates (Wih,Woh,Wuh),
// w7: child0 @ Wfh, w8: child1 @ Wfh.
// ---------------------------------------------------------------------------
__global__ __launch_bounds__(BLOCK) void tree_level_kernel(
    const float* __restrict__ embeds,
    const int* __restrict__ node_words,
    const int* __restrict__ node_labels,
    const __bf16* __restrict__ wpack,
    const float* __restrict__ bix, const float* __restrict__ bih,
    const float* __restrict__ bfx, const float* __restrict__ bfh,
    const float* __restrict__ box_, const float* __restrict__ boh,
    const float* __restrict__ bux, const float* __restrict__ buh,
    const float* __restrict__ Wout, const float* __restrict__ bout,
    const float* __restrict__ hprev, const float* __restrict__ cprev,
    float* __restrict__ hcur, float* __restrict__ ccur,
    float* __restrict__ lossbuf, float* __restrict__ d_out,
    int off, int n, int has_children, int is_root) {
  __shared__ float xT[TM][LSTRIDE];    // x (then reused as h) tile
  __shared__ float hsT[TM][LSTRIDE];   // child-h sum
  __shared__ float ch0T[TM][LSTRIDE];  // child0 h
  __shared__ float ch1T[TM][LSTRIDE];  // child1 h
  __shared__ float outT[NWAVE][TM][LSTRIDE];
  __shared__ float lg[TM][8];

  const int tid = threadIdx.x;
  const int tile0 = blockIdx.x * TM;

  // ---- stage inputs into LDS ----
  for (int idx = tid; idx < TM * HID; idx += BLOCK) {
    int m = idx >> 7, d = idx & 127;
    int node = tile0 + m;
    int cn = node < n ? node : (n - 1);   // clamp for partial top tiles
    int word = node_words[off + cn];
    xT[m][d] = embeds[(size_t)word * HID + d];
    if (has_children) {
      int r0 = cn * 2;
      float h0 = hprev[(size_t)r0 * HID + d];
      float h1 = hprev[(size_t)(r0 + 1) * HID + d];
      ch0T[m][d] = h0;
      ch1T[m][d] = h1;
      hsT[m][d] = h0 + h1;
    }
  }
  __syncthreads();

  // ---- WMMA phase: wave w computes one 16x128 product ----
  const int w = tid >> 5;
  const int lane = tid & 31;
  if (has_children || w < 4) {
    const float(*At)[LSTRIDE] = (w < 4) ? xT : (w < 7) ? hsT : (w == 7) ? ch0T : ch1T;
    const int widx = (w <= 7) ? w : 7;
    const int mrow = lane & 15;
    const int khalf = (lane >> 4) * 16;  // A layout: lanes 0-15 -> K {0..15}; 16-31 -> {16..31}

    v8f acc[8];
#pragma unroll
    for (int jt = 0; jt < 8; ++jt) {
      v8f z = {0.f, 0.f, 0.f, 0.f, 0.f, 0.f, 0.f, 0.f};
      acc[jt] = z;
    }

    for (int kt = 0; kt < 4; ++kt) {
      const int kb = kt * 32 + khalf;
      v16bf a;
#pragma unroll
      for (int h2 = 0; h2 < 8; ++h2) {
        a[h2]     = (__bf16)At[mrow][kb + h2];
        a[h2 + 8] = (__bf16)At[mrow][kb + 8 + h2];
      }
#pragma unroll
      for (int jt = 0; jt < 8; ++jt) {
        const v16bf b = *(const v16bf*)(wpack + (size_t)widx * (HID * HID) +
                                        (size_t)(jt * 4 + kt) * 512 + lane * 16);
        acc[jt] = __builtin_amdgcn_wmma_f32_16x16x32_bf16(
            false, a, false, b, (short)0, acc[jt], false, false);
      }
    }
    // C/D layout: VGPR r -> M = r + (lane<16 ? 0 : 8), N = lane&15
#pragma unroll
    for (int r = 0; r < 8; ++r) {
      const int row = (lane < 16) ? r : (r + 8);
#pragma unroll
      for (int jt = 0; jt < 8; ++jt) outT[w][row][jt * 16 + mrow] = acc[jt][r];
    }
  }
  __syncthreads();

  // ---- elementwise gates, write h/c ----
  for (int idx = tid; idx < TM * HID; idx += BLOCK) {
    int m = idx >> 7, d = idx & 127;
    int node = tile0 + m;
    bool valid = node < n;
    float xi = outT[0][m][d], xo = outT[1][m][d], xu = outT[2][m][d], xf = outT[3][m][d];
    float hi = 0.f, ho = 0.f, hu = 0.f, cagg = 0.f;
    if (has_children) {
      hi = outT[4][m][d];
      ho = outT[5][m][d];
      hu = outT[6][m][d];
      int cn = valid ? node : (n - 1);
      float cc0 = cprev[(size_t)(cn * 2) * HID + d];
      float cc1 = cprev[(size_t)(cn * 2 + 1) * HID + d];
      float fb = xf + bfx[d] + bfh[d];
      float f0 = sigmoidf_(fb + outT[7][m][d]);
      float f1 = sigmoidf_(fb + outT[8][m][d]);
      cagg = f0 * cc0 + f1 * cc1;
    }
    float iv = sigmoidf_(xi + hi + bix[d] + bih[d]);
    float ov = sigmoidf_(xo + ho + box_[d] + boh[d]);
    float uv = tanhf(xu + hu + bux[d] + buh[d]);
    float c = iv * uv + cagg;
    float h = ov * tanhf(c);
    if (valid) {
      hcur[(size_t)node * HID + d] = h;
      ccur[(size_t)node * HID + d] = c;
    }
    xT[m][d] = h;  // reuse x tile as h tile for the output head
  }
  __syncthreads();

  // ---- output head: logits = h @ Wout.T + bout (5 classes) ----
  if (tid < TM * 5) {
    int m = tid / 5, j = tid % 5;
    if (tile0 + m < n) {
      float s = bout[j];
      for (int d = 0; d < HID; ++d) s += xT[m][d] * Wout[j * HID + d];
      lg[m][j] = s;
    }
  }
  __syncthreads();

  // ---- log-softmax + per-node loss (deterministic: no float atomics) ----
  if (tid < TM) {
    int m = tid, node = tile0 + m;
    if (node < n) {
      float mx = lg[m][0];
      for (int j = 1; j < 5; ++j) mx = fmaxf(mx, lg[m][j]);
      float se = 0.f;
      for (int j = 0; j < 5; ++j) se += expf(lg[m][j] - mx);
      float lse = mx + logf(se);
      int lab = node_labels[off + node];
      lossbuf[off + node] = -(lg[m][lab] - lse);
      if (is_root && node == 0) {
        for (int j = 0; j < 5; ++j) d_out[j] = lg[m][j] - lse;
      }
    }
  }
}

// Fixed-order reduction of per-node losses -> d_out[5]. Deterministic.
__global__ __launch_bounds__(256) void loss_reduce_kernel(
    const float* __restrict__ lossbuf, float* __restrict__ d_out, int nnodes) {
  __shared__ float part[256];
  float s = 0.f;
  for (int i = threadIdx.x; i < nnodes; i += 256) s += lossbuf[i];
  part[threadIdx.x] = s;
  __syncthreads();
  for (int st = 128; st > 0; st >>= 1) {
    if ((int)threadIdx.x < st) part[threadIdx.x] += part[threadIdx.x + st];
    __syncthreads();
  }
  if (threadIdx.x == 0) d_out[5] = part[0];
}

extern "C" void kernel_launch(void* const* d_in, const int* in_sizes, int n_in,
                              void* d_out, int out_size, void* d_ws, size_t ws_size,
                              hipStream_t stream) {
  const float* embeds      = (const float*)d_in[0];
  const int*   node_words  = (const int*)d_in[1];
  const int*   node_labels = (const int*)d_in[2];
  // d_in[3]=depth(18), d_in[4]=branching(2): fixed by problem constants.
  const float* Wix = (const float*)d_in[5];  const float* bix = (const float*)d_in[6];
  const float* Wih = (const float*)d_in[7];  const float* bih = (const float*)d_in[8];
  const float* Wfx = (const float*)d_in[9];  const float* bfx = (const float*)d_in[10];
  const float* Wfh = (const float*)d_in[11]; const float* bfh = (const float*)d_in[12];
  const float* Wox = (const float*)d_in[13]; const float* box_ = (const float*)d_in[14];
  const float* Woh = (const float*)d_in[15]; const float* boh = (const float*)d_in[16];
  const float* Wux = (const float*)d_in[17]; const float* bux = (const float*)d_in[18];
  const float* Wuh = (const float*)d_in[19]; const float* buh = (const float*)d_in[20];
  const float* Wout = (const float*)d_in[21]; const float* bout = (const float*)d_in[22];
  float* out = (float*)d_out;

  // workspace layout
  char* ws = (char*)d_ws;
  __bf16* wpack  = (__bf16*)ws;                       // 8*128*128*2 = 256 KB
  float*  lossbuf = (float*)(ws + (256 << 10));       // NNODES f32 ~ 1 MB
  size_t base = (256 << 10) + (1 << 20);
  const size_t bufbytes = (size_t)(1 << (TREE_DEPTH - 1)) * HID * sizeof(float); // 64 MB
  float* hbuf[2] = {(float*)(ws + base),                (float*)(ws + base + bufbytes)};
  float* cbuf[2] = {(float*)(ws + base + 2 * bufbytes), (float*)(ws + base + 3 * bufbytes)};

  pack_weights_kernel<<<8, 256, 0, stream>>>(Wix, Wox, Wux, Wfx, Wih, Woh, Wuh, Wfh, wpack);

  for (int l = TREE_DEPTH - 1; l >= 0; --l) {
    int n = 1 << l;
    int off = (1 << l) - 1;
    int cur = l & 1, prev = cur ^ 1;
    int has_children = (l != TREE_DEPTH - 1);
    int is_root = (l == 0);
    int nblocks = (n + TM - 1) / TM;
    tree_level_kernel<<<nblocks, BLOCK, 0, stream>>>(
        embeds, node_words, node_labels, wpack,
        bix, bih, bfx, bfh, box_, boh, bux, buh, Wout, bout,
        hbuf[prev], cbuf[prev], hbuf[cur], cbuf[cur],
        lossbuf, out, off, n, has_children, is_root);
  }

  loss_reduce_kernel<<<1, 256, 0, stream>>>(lossbuf, out, NNODES);
}
